// NoiseConditionedMoE_59974923321727
// MI455X (gfx1250) — compile-verified, matrix operand order checked by reading, not run on hardware
//
#include <hip/hip_runtime.h>
#include <hip/hip_bf16.h>
#include <math.h>

// Problem constants (match reference)
#define BDIM   8
#define SDIM   1024
#define DDIM   1024
#define HDIM   2048
#define EDIM   8
#define TOPK   2
#define DNOISE 256

typedef __attribute__((ext_vector_type(16))) __bf16 v16bf;
typedef __attribute__((ext_vector_type(8)))  __bf16 v8bf;
typedef __attribute__((ext_vector_type(8)))  float  v8f;
typedef __attribute__((ext_vector_type(4)))  unsigned v4u;
typedef __attribute__((ext_vector_type(8)))  int    v8i;
typedef __attribute__((ext_vector_type(4)))  int    v4i;

#if defined(__has_builtin)
#if __has_builtin(__builtin_amdgcn_tensor_load_to_lds) && \
    __has_builtin(__builtin_amdgcn_s_wait_tensorcnt)
#define HAVE_TDM 1
#endif
#endif

// ---------------------------------------------------------------------------
// Workspace layout (bytes):
//   [0, 64)            : float  norm. top-k weights [b][k]
//   [64, 128)          : int    top-k expert idx   [b][k]
//   [256, +64MiB)      : hidden bf16 [B*K][S][H]            (k-major, WMMA A)
//   next  16MiB        : x bf16      [B*S][D]               (k-major, WMMA A)
//   next  64MiB        : fc_in_w^T bf16  [E][2H][D]         (k-major, WMMA B)
//   next  32MiB        : fc_out_w^T bf16 [E][D][H]          (k-major, WMMA B)
// ---------------------------------------------------------------------------
#define WS_HIDDEN_OFF   ((size_t)256)
#define WS_XBF_OFF      (WS_HIDDEN_OFF + (size_t)BDIM * TOPK * SDIM * HDIM * 2)
#define WS_WINT_OFF     (WS_XBF_OFF + (size_t)BDIM * SDIM * DDIM * 2)
#define WS_WOUTT_OFF    (WS_WINT_OFF + (size_t)EDIM * 2 * HDIM * DDIM * 2)

__device__ __forceinline__ v8f wmma_bf16(v16bf a, v16bf b, v8f c) {
  return __builtin_amdgcn_wmma_f32_16x16x32_bf16(
      false, a, false, b, (short)0, c, false, false);
}

// Fragment load from k-major bf16 memory (global or LDS; A and B operands):
// lane-half r = lane&15 selects the row (A: M) / column (B: N); K-halves
// {0..7,16..23} vs {8..15,24..31} are contiguous 16-byte runs per lane.
__device__ __forceinline__ v16bf ldfrag(const __bf16* __restrict__ base,
                                        int ld, int lane) {
  const int r     = lane & 15;
  const int khalf = (lane >> 4) & 1;
  const __bf16* p = base + (size_t)r * ld + khalf * 8;
  const v8bf lo = *reinterpret_cast<const v8bf*>(p);
  const v8bf hi = *reinterpret_cast<const v8bf*>(p + 16);
  return __builtin_shufflevector(lo, hi, 0, 1, 2, 3, 4, 5, 6, 7,
                                 8, 9, 10, 11, 12, 13, 14, 15);
}

#ifdef HAVE_TDM
// ---------------------------------------------------------------------------
// TDM: issue a 2D tensor-tile load (bf16) global -> LDS.
// Descriptor packing per CDNA5 ISA 8.3/8.4 (D# group0 / group1), 2D tile.
// 6-arg builtin: (uint32x4 g0, int32x8 g1, int32x4, int32x4, int32x8, cpol)
// ---------------------------------------------------------------------------
__device__ __forceinline__ void tdm_load_2d(unsigned lds_byte_addr,
                                            const void* gptr,
                                            unsigned tile0, unsigned tile1,
                                            unsigned long long stride0) {
  const unsigned long long ga = (unsigned long long)gptr;
  v4u g0;
  g0[0] = 1u;                                            // count=1 (valid)
  g0[1] = lds_byte_addr;                                 // LDS dest (bytes)
  g0[2] = (unsigned)(ga & 0xFFFFFFFFull);                // global_addr[31:0]
  g0[3] = (unsigned)((ga >> 32) & 0x01FFFFFFull)         // global_addr[56:32]
          | (2u << 30);                                  // type=2 ("image")
  v8i g1;
  g1[0] = (int)(1u << 16);                               // data_size=1 (2B)
  g1[1] = (int)((tile0 & 0xFFFFu) << 16);                // tensor_dim0[15:0]
  g1[2] = (int)(((tile0 >> 16) & 0xFFFFu) |
                ((tile1 & 0xFFFFu) << 16));              // dim0 hi | dim1 lo
  g1[3] = (int)(((tile1 >> 16) & 0xFFFFu) |
                ((tile0 & 0xFFFFu) << 16));              // dim1 hi | tile_dim0
  g1[4] = (int)(tile1 & 0xFFFFu);                        // tile_dim1 (2D)
  g1[5] = (int)(unsigned)(stride0 & 0xFFFFFFFFull);      // dim0_stride[31:0]
  g1[6] = (int)(unsigned)((stride0 >> 32) & 0xFFFFull);  // dim0_stride[47:32]
  g1[7] = 0;
  const v4i z4 = {0, 0, 0, 0};
  const v8i z8 = {0, 0, 0, 0, 0, 0, 0, 0};
  __builtin_amdgcn_tensor_load_to_lds(g0, g1, z4, z4, z8, 0);
}

__device__ __forceinline__ unsigned lds_off(const void* p) {
  // LDS aperture: generic address low 32 bits == LDS byte address.
  return (unsigned)(unsigned long long)p;
}
#endif  // HAVE_TDM

// ---------------------------------------------------------------------------
// Kernel 0a: elementwise fp32 -> bf16 (x), 8 elements per thread, b128 I/O.
// ---------------------------------------------------------------------------
__global__ __launch_bounds__(256)
void cvt_bf16_kernel(const float* __restrict__ src, __bf16* __restrict__ dst,
                     int n8) {
  const int i = blockIdx.x * 256 + threadIdx.x;
  if (i < n8) {
    const float4* s = reinterpret_cast<const float4*>(src);
    const float4 a = s[2 * i];
    const float4 b = s[2 * i + 1];
    v8bf o;
    o[0] = (__bf16)a.x; o[1] = (__bf16)a.y; o[2] = (__bf16)a.z; o[3] = (__bf16)a.w;
    o[4] = (__bf16)b.x; o[5] = (__bf16)b.y; o[6] = (__bf16)b.z; o[7] = (__bf16)b.w;
    *reinterpret_cast<v8bf*>(dst + (size_t)8 * i) = o;
  }
}

// ---------------------------------------------------------------------------
// Kernel 0b: transpose + convert: src fp32 [K][N] -> dst bf16 [N][K], per
// blockIdx.z slice (expert). 64x64 tiles through padded LDS.
// ---------------------------------------------------------------------------
__global__ __launch_bounds__(256)
void transpose_cvt_kernel(const float* __restrict__ src, __bf16* __restrict__ dst,
                          int K, int N) {
  __shared__ float tile[64][65];
  const size_t eoff = (size_t)blockIdx.z * K * N;
  const int n0 = blockIdx.x * 64;
  const int k0 = blockIdx.y * 64;
  const int c  = threadIdx.x & 63;
  const int r4 = threadIdx.x >> 6;
#pragma unroll
  for (int i = 0; i < 16; ++i) {
    const int r = r4 + 4 * i;
    tile[r][c] = src[eoff + (size_t)(k0 + r) * N + n0 + c];
  }
  __syncthreads();
#pragma unroll
  for (int i = 0; i < 16; ++i) {
    const int r = r4 + 4 * i;                   // output row = n index
    dst[eoff + (size_t)(n0 + r) * K + k0 + c] = (__bf16)tile[c][r];
  }
}

// ---------------------------------------------------------------------------
// Kernel 1: router (logits -> softmax -> top-2 -> normalized weights)
// ---------------------------------------------------------------------------
__global__ __launch_bounds__(64)
void moe_router_kernel(const float* __restrict__ noise,
                       const float* __restrict__ route_w,
                       float* __restrict__ out_logits,
                       float* __restrict__ out_probs,
                       float* __restrict__ out_idx,
                       float* __restrict__ out_w,
                       float* __restrict__ ws_w,
                       int*   __restrict__ ws_idx) {
  __shared__ float lg[BDIM][EDIM];
  const int tid = threadIdx.x;
  const int b = tid >> 3;
  const int e = tid & 7;
  float acc = 0.f;
#pragma unroll 4
  for (int i = 0; i < DNOISE; ++i)
    acc += noise[b * DNOISE + i] * route_w[i * EDIM + e];
  lg[b][e] = acc;
  out_logits[b * EDIM + e] = acc;   // TEMP == 1.0
  __syncthreads();
  if (e == 0) {
    float m = lg[b][0];
#pragma unroll
    for (int j = 1; j < EDIM; ++j) m = fmaxf(m, lg[b][j]);
    float p[EDIM];
    float s = 0.f;
#pragma unroll
    for (int j = 0; j < EDIM; ++j) { p[j] = expf(lg[b][j] - m); s += p[j]; }
    const float inv = 1.f / s;
#pragma unroll
    for (int j = 0; j < EDIM; ++j) { p[j] *= inv; out_probs[b * EDIM + j] = p[j]; }
    int i0 = 0;
#pragma unroll
    for (int j = 1; j < EDIM; ++j) if (p[j] > p[i0]) i0 = j;
    int i1 = (i0 == 0) ? 1 : 0;
#pragma unroll
    for (int j = 0; j < EDIM; ++j) if (j != i0 && p[j] > p[i1]) i1 = j;
    const float w0 = p[i0], w1 = p[i1];
    const float denom = fmaxf(w0 + w1, 1e-8f);
    const float n0 = w0 / denom, n1 = w1 / denom;
    out_idx[b * TOPK + 0] = (float)i0;
    out_idx[b * TOPK + 1] = (float)i1;
    out_w[b * TOPK + 0] = n0;
    out_w[b * TOPK + 1] = n1;
    ws_w[b * TOPK + 0] = n0;
    ws_w[b * TOPK + 1] = n1;
    ws_idx[b * TOPK + 0] = i0;
    ws_idx[b * TOPK + 1] = i1;
  }
}

// ---------------------------------------------------------------------------
// Kernel 2: fc_in + SwiGLU, selected experts only.
// Grid: (H/64, S/64, B*K). 8 waves (4x2); wave = 16 rows x 32 hidden cols.
// TDM producer waves 0/1/2 stream A / W_value / W_gate 64x64 bf16 slabs into
// double-buffered LDS; consumers issue all 20 ds_load_b128 of a buffer before
// the 8 WMMAs so loads overlap matrix ops.
// ---------------------------------------------------------------------------
__global__ __launch_bounds__(256)
void moe_fc_in_kernel(const __bf16* __restrict__ xbf,
                      const __bf16* __restrict__ w_in_t,
                      const float*  __restrict__ fc_in_b,
                      const int*    __restrict__ ws_idx,
                      __bf16* __restrict__ hidden) {
  const int lane = threadIdx.x & 31;
  const int wv = threadIdx.x >> 5;
  const int wvu = __builtin_amdgcn_readfirstlane(wv);   // SGPR wave id
  const int wm = wv >> 1;                 // 0..3
  const int wn = wv & 1;                  // 0..1
  const int bk = blockIdx.z;              // 0..15
  const int b  = bk >> 1;
  const int e  = ws_idx[bk];
  const int s0b = blockIdx.y * 64;
  const int h0b = blockIdx.x * 64;
  const int s0 = s0b + wm * 16;
  const int h0 = h0b + wn * 32;

  const __bf16* xb = xbf + (size_t)(b * SDIM + s0b) * DDIM;
  const __bf16* wvp = w_in_t + (size_t)e * (2 * HDIM) * DDIM + (size_t)h0b * DDIM;
  const __bf16* wgp = w_in_t + (size_t)e * (2 * HDIM) * DDIM +
                      (size_t)(HDIM + h0b) * DDIM;

  v8f accv0 = {}, accv1 = {}, accg0 = {}, accg1 = {};

#ifdef HAVE_TDM
  __shared__ __bf16 sA[2][64][64];        // [buf][s-row][k]
  __shared__ __bf16 sB[2][2][64][64];     // [buf][val/gate][h-col][k]

  // Prologue: fill buffer 0 (kb = 0).
  if (wvu == 0)      tdm_load_2d(lds_off(&sA[0][0][0]), xb,  64, 64, DDIM);
  else if (wvu == 1) tdm_load_2d(lds_off(&sB[0][0][0][0]), wvp, 64, 64, DDIM);
  else if (wvu == 2) tdm_load_2d(lds_off(&sB[0][1][0][0]), wgp, 64, 64, DDIM);

  int buf = 0;
  for (int kb = 0; kb < DDIM; kb += 64, buf ^= 1) {
    __builtin_amdgcn_s_wait_tensorcnt(0);   // producer waves: current buf done
    __syncthreads();                        // publish LDS; prior reads of buf^1 done
    if (kb + 64 < DDIM) {                   // stream next slabs into other buffer
      if (wvu == 0)      tdm_load_2d(lds_off(&sA[buf ^ 1][0][0]), xb + kb + 64, 64, 64, DDIM);
      else if (wvu == 1) tdm_load_2d(lds_off(&sB[buf ^ 1][0][0][0]), wvp + kb + 64, 64, 64, DDIM);
      else if (wvu == 2) tdm_load_2d(lds_off(&sB[buf ^ 1][1][0][0]), wgp + kb + 64, 64, 64, DDIM);
    }
    // Issue all fragment loads for both 32-K substeps, then all 8 WMMAs.
    v16bf aA  = ldfrag(&sA[buf][wm * 16][0],           64, lane);
    v16bf vA0 = ldfrag(&sB[buf][0][wn * 32][0],        64, lane);
    v16bf vA1 = ldfrag(&sB[buf][0][wn * 32 + 16][0],   64, lane);
    v16bf gA0 = ldfrag(&sB[buf][1][wn * 32][0],        64, lane);
    v16bf gA1 = ldfrag(&sB[buf][1][wn * 32 + 16][0],   64, lane);
    v16bf aB  = ldfrag(&sA[buf][wm * 16][32],          64, lane);
    v16bf vB0 = ldfrag(&sB[buf][0][wn * 32][32],       64, lane);
    v16bf vB1 = ldfrag(&sB[buf][0][wn * 32 + 16][32],  64, lane);
    v16bf gB0 = ldfrag(&sB[buf][1][wn * 32][32],       64, lane);
    v16bf gB1 = ldfrag(&sB[buf][1][wn * 32 + 16][32],  64, lane);
    accv0 = wmma_bf16(aA, vA0, accv0);
    accv1 = wmma_bf16(aA, vA1, accv1);
    accg0 = wmma_bf16(aA, gA0, accg0);
    accg1 = wmma_bf16(aA, gA1, accg1);
    accv0 = wmma_bf16(aB, vB0, accv0);
    accv1 = wmma_bf16(aB, vB1, accv1);
    accg0 = wmma_bf16(aB, gB0, accg0);
    accg1 = wmma_bf16(aB, gB1, accg1);
  }
#else
  const __bf16* xw = xb + (size_t)(wm * 16) * DDIM;
  const __bf16* pv0 = wvp + (size_t)(wn * 32) * DDIM;
  const __bf16* pv1 = wvp + (size_t)(wn * 32 + 16) * DDIM;
  const __bf16* pg0 = wgp + (size_t)(wn * 32) * DDIM;
  const __bf16* pg1 = wgp + (size_t)(wn * 32 + 16) * DDIM;
  for (int kb = 0; kb < DDIM; kb += 32) {
    __builtin_prefetch(pv0 + kb + 64, 0, 1);
    v16bf a   = ldfrag(xw + kb, DDIM, lane);
    v16bf bv0 = ldfrag(pv0 + kb, DDIM, lane);
    v16bf bv1 = ldfrag(pv1 + kb, DDIM, lane);
    v16bf bg0 = ldfrag(pg0 + kb, DDIM, lane);
    v16bf bg1 = ldfrag(pg1 + kb, DDIM, lane);
    accv0 = wmma_bf16(a, bv0, accv0);
    accv1 = wmma_bf16(a, bv1, accv1);
    accg0 = wmma_bf16(a, bg0, accg0);
    accg1 = wmma_bf16(a, bg1, accg1);
  }
#endif

  const int n  = lane & 15;
  const int hi = lane >> 4;
#pragma unroll
  for (int t = 0; t < 2; ++t) {
    const int col = h0 + 16 * t + n;
    const float bv = fc_in_b[(size_t)e * (2 * HDIM) + col];
    const float bg = fc_in_b[(size_t)e * (2 * HDIM) + HDIM + col];
#pragma unroll
    for (int r = 0; r < 8; ++r) {
      const float v = (t == 0 ? accv0[r] : accv1[r]) + bv;
      const float g = (t == 0 ? accg0[r] : accg1[r]) + bg;
      const float hval = v * (g / (1.f + expf(-g)));   // value * silu(gate)
      const int row = s0 + r + 8 * hi;
      hidden[(size_t)(bk * SDIM + row) * HDIM + col] = (__bf16)hval;
    }
  }
}

// ---------------------------------------------------------------------------
// Kernel 3: fc_out, weighted sum of 2 selected experts + weighted bias.
// Grid: (D/128, S/64, B). 8 waves (4x2); wave = 16 rows x 64 cols (4 tiles).
// TDM path: wave 0 streams the hidden slab (64x64), waves 1/2 each stream
// half of the 128-row W_out slab into double-buffered LDS.
// ---------------------------------------------------------------------------
__global__ __launch_bounds__(256)
void moe_fc_out_kernel(const __bf16* __restrict__ hidden,
                       const __bf16* __restrict__ w_out_t,
                       const float*  __restrict__ fc_out_b,
                       const int*    __restrict__ ws_idx,
                       const float*  __restrict__ ws_w,
                       float* __restrict__ out) {
  const int lane = threadIdx.x & 31;
  const int wv = threadIdx.x >> 5;
  const int wvu = __builtin_amdgcn_readfirstlane(wv);
  const int wm = wv >> 1;                 // 0..3
  const int wn = wv & 1;                  // 0..1
  const int b  = blockIdx.z;
  const int s0b = blockIdx.y * 64;
  const int d0b = blockIdx.x * 128;
  const int s0 = s0b + wm * 16;
  const int d0 = d0b + wn * 64;

  const int n  = lane & 15;
  const int hi = lane >> 4;

  v8f fin[4] = {};
  float bcol[4] = {0.f, 0.f, 0.f, 0.f};

#ifdef HAVE_TDM
  __shared__ __bf16 sA[2][64][64];        // [buf][s-row][k]
  __shared__ __bf16 sB[2][128][64];       // [buf][d-col][k]
#endif

#pragma unroll 1
  for (int kk = 0; kk < TOPK; ++kk) {
    const int   e  = ws_idx[b * TOPK + kk];
    const float wk = ws_w[b * TOPK + kk];
    const __bf16* hb = hidden + (size_t)((b * TOPK + kk) * SDIM + s0b) * HDIM;
    const __bf16* wo = w_out_t + (size_t)e * DDIM * HDIM + (size_t)d0b * HDIM;

    v8f acc[4] = {};
#ifdef HAVE_TDM
    __syncthreads();                      // previous pass done reading LDS
    if (wvu == 0)      tdm_load_2d(lds_off(&sA[0][0][0]), hb, 64, 64, HDIM);
    else if (wvu == 1) tdm_load_2d(lds_off(&sB[0][0][0]), wo, 64, 64, HDIM);
    else if (wvu == 2) tdm_load_2d(lds_off(&sB[0][64][0]),
                                   wo + (size_t)64 * HDIM, 64, 64, HDIM);
    int buf = 0;
    for (int kb = 0; kb < HDIM; kb += 64, buf ^= 1) {
      __builtin_amdgcn_s_wait_tensorcnt(0);
      __syncthreads();
      if (kb + 64 < HDIM) {
        if (wvu == 0)      tdm_load_2d(lds_off(&sA[buf ^ 1][0][0]), hb + kb + 64, 64, 64, HDIM);
        else if (wvu == 1) tdm_load_2d(lds_off(&sB[buf ^ 1][0][0]), wo + kb + 64, 64, 64, HDIM);
        else if (wvu == 2) tdm_load_2d(lds_off(&sB[buf ^ 1][64][0]),
                                       wo + (size_t)64 * HDIM + kb + 64, 64, 64, HDIM);
      }
      v16bf aA = ldfrag(&sA[buf][wm * 16][0], 64, lane);
      v16bf bA0 = ldfrag(&sB[buf][wn * 64][0],      64, lane);
      v16bf bA1 = ldfrag(&sB[buf][wn * 64 + 16][0], 64, lane);
      v16bf bA2 = ldfrag(&sB[buf][wn * 64 + 32][0], 64, lane);
      v16bf bA3 = ldfrag(&sB[buf][wn * 64 + 48][0], 64, lane);
      v16bf aB = ldfrag(&sA[buf][wm * 16][32], 64, lane);
      v16bf bB0 = ldfrag(&sB[buf][wn * 64][32],      64, lane);
      v16bf bB1 = ldfrag(&sB[buf][wn * 64 + 16][32], 64, lane);
      v16bf bB2 = ldfrag(&sB[buf][wn * 64 + 32][32], 64, lane);
      v16bf bB3 = ldfrag(&sB[buf][wn * 64 + 48][32], 64, lane);
      acc[0] = wmma_bf16(aA, bA0, acc[0]);
      acc[1] = wmma_bf16(aA, bA1, acc[1]);
      acc[2] = wmma_bf16(aA, bA2, acc[2]);
      acc[3] = wmma_bf16(aA, bA3, acc[3]);
      acc[0] = wmma_bf16(aB, bB0, acc[0]);
      acc[1] = wmma_bf16(aB, bB1, acc[1]);
      acc[2] = wmma_bf16(aB, bB2, acc[2]);
      acc[3] = wmma_bf16(aB, bB3, acc[3]);
    }
#else
    const __bf16* hw = hb + (size_t)(wm * 16) * HDIM;
    for (int kb = 0; kb < HDIM; kb += 32) {
      __builtin_prefetch(wo + (size_t)(wn * 64) * HDIM + kb + 64, 0, 1);
      v16bf a = ldfrag(hw + kb, HDIM, lane);
#pragma unroll
      for (int t = 0; t < 4; ++t) {
        v16bf bt = ldfrag(wo + (size_t)(wn * 64 + 16 * t) * HDIM + kb, HDIM, lane);
        acc[t] = wmma_bf16(a, bt, acc[t]);
      }
    }
#endif
#pragma unroll
    for (int t = 0; t < 4; ++t) {
#pragma unroll
      for (int r = 0; r < 8; ++r) fin[t][r] += wk * acc[t][r];
      bcol[t] += wk * fc_out_b[(size_t)e * DDIM + d0 + 16 * t + n];
    }
  }

#pragma unroll
  for (int t = 0; t < 4; ++t)
#pragma unroll
    for (int r = 0; r < 8; ++r) {
      const int row = s0 + r + 8 * hi;
      out[(size_t)(b * SDIM + row) * DDIM + d0 + 16 * t + n] = fin[t][r] + bcol[t];
    }
}

// ---------------------------------------------------------------------------
extern "C" void kernel_launch(void* const* d_in, const int* in_sizes, int n_in,
                              void* d_out, int out_size, void* d_ws, size_t ws_size,
                              hipStream_t stream) {
  (void)in_sizes; (void)n_in; (void)out_size; (void)ws_size;

  const float* x        = (const float*)d_in[0];  // [B,S,D]
  const float* noise    = (const float*)d_in[1];  // [B,256]
  const float* route_w  = (const float*)d_in[2];  // [256,E]
  const float* fc_in_w  = (const float*)d_in[3];  // [E,D,2H]
  const float* fc_in_b  = (const float*)d_in[4];  // [E,2H]
  const float* fc_out_w = (const float*)d_in[5];  // [E,H,D]
  const float* fc_out_b = (const float*)d_in[6];  // [E,D]

  float* out      = (float*)d_out;
  float* mixed    = out;                                   // [B,S,D]
  float* o_logits = out + (size_t)BDIM * SDIM * DDIM;      // [B,E]
  float* o_probs  = o_logits + BDIM * EDIM;                // [B,E]
  float* o_idx    = o_probs + BDIM * EDIM;                 // [B,K] (as float)
  float* o_w      = o_idx + BDIM * TOPK;                   // [B,K]

  float*  ws_w    = (float*)d_ws;                          // [B,K]
  int*    ws_idx  = (int*)((char*)d_ws + 64);              // [B,K]
  __bf16* hidden  = (__bf16*)((char*)d_ws + WS_HIDDEN_OFF);
  __bf16* xbf     = (__bf16*)((char*)d_ws + WS_XBF_OFF);
  __bf16* w_in_t  = (__bf16*)((char*)d_ws + WS_WINT_OFF);
  __bf16* w_out_t = (__bf16*)((char*)d_ws + WS_WOUTT_OFF);

  // 1) router (independent of conversions)
  moe_router_kernel<<<1, 64, 0, stream>>>(noise, route_w, o_logits, o_probs,
                                          o_idx, o_w, ws_w, ws_idx);

  // 2) one-time precision/layout conversions
  {
    const int n8 = (BDIM * SDIM * DDIM) / 8;               // 1,048,576
    cvt_bf16_kernel<<<n8 / 256, 256, 0, stream>>>(x, xbf, n8);
  }
  transpose_cvt_kernel<<<dim3((2 * HDIM) / 64, DDIM / 64, EDIM), 256, 0, stream>>>(
      fc_in_w, w_in_t, DDIM, 2 * HDIM);
  transpose_cvt_kernel<<<dim3(DDIM / 64, HDIM / 64, EDIM), 256, 0, stream>>>(
      fc_out_w, w_out_t, HDIM, DDIM);

  // 3) expert GEMMs (top-2 only)
  moe_fc_in_kernel<<<dim3(HDIM / 64, SDIM / 64, BDIM * TOPK), 256, 0, stream>>>(
      xbf, w_in_t, fc_in_b, ws_idx, hidden);

  moe_fc_out_kernel<<<dim3(DDIM / 128, SDIM / 64, BDIM), 256, 0, stream>>>(
      hidden, w_out_t, fc_out_b, ws_idx, ws_w, mixed);
}
